// VGRUModel_8796093022919
// MI455X (gfx1250) — matrix operand order repaired and checked by reading, not compiled, not run on hardware
//
#include <hip/hip_runtime.h>
#include <hip/hip_bf16.h>
#include <math.h>

// ---------- problem constants ----------
#define NUM_HS 13
#define BB 64
#define TT 128
#define HH 768
#define UU 32
#define DD 64        // 2*U
#define NN 130       // D*LABELS + LABELS
#define SMC 1000

typedef __attribute__((ext_vector_type(16))) _Float16 v16h;
typedef __attribute__((ext_vector_type(8)))  _Float16 v8h;
typedef __attribute__((ext_vector_type(8)))  float    v8f;
typedef unsigned int  u32x4 __attribute__((ext_vector_type(4)));
typedef int           i32x4 __attribute__((ext_vector_type(4)));
typedef int           i32x8 __attribute__((ext_vector_type(8)));

union AFrag { v16h v; v8h p[2]; _Float16 h[16]; };
union CFrag { v8f  v; float f[8]; };

// ---------------------------------------------------------------
// CDNA5 async global->LDS DMA (ASYNCcnt-tracked), per ISA 10.7 / 15.18.3.
// One 16-byte transfer per active lane, from global address vaddr to LDS address vdst.
__device__ __forceinline__ void async_b128(const void* g, void* l) {
    unsigned lds_off = (unsigned)(uintptr_t)l;   // generic LDS ptr: low 32 bits = LDS addr
    asm volatile("global_load_async_to_lds_b128 %0, %1, off"
                 :: "v"(lds_off), "v"(g) : "memory");
}
__device__ __forceinline__ void wait_async0() {
    asm volatile("s_wait_asynccnt 0x0" ::: "memory");
}

// ---------------------------------------------------------------
// CDNA5 Tensor Data Mover: one-shot 1D DMA global->LDS (TENSORcnt-tracked).
// D# per ISA 8.3/8.4: group0 = {count=1, lds_addr, global_addr, type=2};
// group1 = {mask=0, data_size=4B, tensor_dim0=elems, tensor_dim1=1,
//           tile_dim0=elems (<=65535), stride0=elems}.
#if __has_builtin(__builtin_amdgcn_tensor_load_to_lds)
#define HAVE_TDM 1
__device__ __forceinline__ void tdm_load_1d(const void* gsrc, void* ldst, unsigned elems4B) {
    unsigned long long ga = (unsigned long long)(uintptr_t)gsrc;
    unsigned lds = (unsigned)(uintptr_t)ldst;
    u32x4 g0;
    g0[0] = 1u;                                                   // count=1, user mode
    g0[1] = lds;                                                  // lds_addr (bytes)
    g0[2] = (unsigned)ga;                                         // global_addr[31:0]
    g0[3] = (unsigned)((ga >> 32) & 0x01FFFFFFull) | (2u << 30);  // addr[56:32] | type=2
    i32x8 g1;
    g1[0] = (int)(2u << 16);                      // data_size=2 (4 bytes), no pad/iter
    g1[1] = (int)((elems4B & 0xFFFFu) << 16);     // tensor_dim0[15:0]
    g1[2] = (int)((elems4B >> 16) | (1u << 16));  // tensor_dim0[31:16] | tensor_dim1=1
    g1[3] = (int)(elems4B << 16);                 // tile_dim0 = elems
    g1[4] = 0;                                    // tile_dim1/2 unused
    g1[5] = (int)elems4B;                         // tensor_dim0_stride[31:0]
    g1[6] = 0;
    g1[7] = 0;
    i32x4 z4 = {0, 0, 0, 0};
#if __clang_major__ >= 23
    i32x8 z8 = {0, 0, 0, 0, 0, 0, 0, 0};
    __builtin_amdgcn_tensor_load_to_lds(g0, g1, z4, z4, z8, 0);
#else
    __builtin_amdgcn_tensor_load_to_lds(g0, g1, z4, z4, 0);
#endif
}
#else
#define HAVE_TDM 0
#endif

// ---------------------------------------------------------------
// zero-fill
__global__ void zero_f32(float* __restrict__ p, int n) {
    int i = blockIdx.x * blockDim.x + threadIdx.x;
    if (i < n) p[i] = 0.0f;
}

// ---------------------------------------------------------------
// Pack a row-major fp32 weight W[K][Ncols] into the wave32 WMMA B-fragment
// layout for v_wmma_f32_16x16x32_f16 (ISA 7.12.2):
//   lane<16 : col = nt*16+lane,      K = kc*32 + {0..7, 16..23}
//   lane>=16: col = nt*16+(lane-16), K = kc*32 + {8..15, 24..31}
// out[((nt*nKchunks + kc)*32 + lane)*16 + j]
__global__ void pack_b_f16(const float* __restrict__ W, _Float16* __restrict__ outp,
                           int Ncols, int nKchunks, int total) {
    int tid = blockIdx.x * blockDim.x + threadIdx.x;
    if (tid >= total) return;
    int lane = tid & 31;
    int kc   = (tid >> 5) % nKchunks;
    int nt   = tid / (32 * nKchunks);
    int col  = nt * 16 + (lane & 15);
    int kb   = kc * 32 + (lane >> 4) * 8;
    _Float16* o = outp + (size_t)tid * 16;
#pragma unroll
    for (int j = 0; j < 16; j++)
        o[j] = (_Float16)W[(size_t)(kb + (j & 7) + (j >> 3) * 16) * Ncols + col];
}

// ---------------------------------------------------------------
// Per-layer fused kernel:
//   stage packed lin_w (96KB) -> LDS via TDM (one tensor_load_to_lds per block)
//   zpre = hs_i @ lin_w            (WMMA f16, f32 accum, K=768, B from LDS)
//   z    = LayerNorm(zpre + lin_b + seq_in)      (zpre/zh overlay the B buffer)
//   xp_f = z @ gru_kf + bi_f ;  xp_b = z @ gru_kb + bi_b   (WMMA, K=64)
// One workgroup handles 128 rows of B*T=8192; 8 waves, 16 rows each.
#define SMEM_BYTES 98304      // 49152 halfs of packed lin_w  (>= 33280 + 18432 overlay)
__global__ __launch_bounds__(256)
void layer_kernel(const float* __restrict__ hs_i, const float* __restrict__ seq_in,
                  const _Float16* __restrict__ lwp, const _Float16* __restrict__ kfp,
                  const _Float16* __restrict__ kbp,
                  const float* __restrict__ linb, const float* __restrict__ bif,
                  const float* __restrict__ bib, const float* __restrict__ lng,
                  const float* __restrict__ lnb,
                  float* __restrict__ xpf, float* __restrict__ xpb) {
    __shared__ __align__(32) unsigned char smem[SMEM_BYTES];
    _Float16* Bl  = (_Float16*)smem;             // packed lin_w, phase 1
    float*    zp  = (float*)smem;                // [128][65] overlay, phase 2
    _Float16* zhp = (_Float16*)(smem + 33280);   // [128][72] overlay, phase 2

    const int t    = threadIdx.x;
    const int w    = t >> 5;
    const int lane = t & 31;
    const int lm   = lane & 15;
    const int lh   = lane >> 4;
    const int mb   = w * 16;
    const int rowbase = blockIdx.x * 128;

    // ---- stage packed B into LDS ----
#if HAVE_TDM
    if (t < 32) {                                // wave 0 issues one TDM DMA
        tdm_load_1d(lwp, smem, SMEM_BYTES / 4);
        __builtin_amdgcn_s_wait_tensorcnt(0);
    }
#else
    {
        const unsigned char* src = (const unsigned char*)lwp;
        for (int i = t; i < SMEM_BYTES / 16; i += 256)
            async_b128(src + (size_t)i * 16, smem + (size_t)i * 16);
        wait_async0();
    }
#endif
    __syncthreads();

    // ---- main GEMM: 16 rows per wave, N=64 (4 n-tiles), K=768 (24 chunks) ----
    CFrag acc[4];
#pragma unroll
    for (int nt = 0; nt < 4; nt++)
#pragma unroll
        for (int j = 0; j < 8; j++) acc[nt].f[j] = 0.0f;

    const float* arow = hs_i + (size_t)(rowbase + mb + lm) * HH;
    for (int kc = 0; kc < 24; kc++) {
        const int k0 = kc * 32 + lh * 8;
        float4 f0 = *(const float4*)(arow + k0);
        float4 f1 = *(const float4*)(arow + k0 + 4);
        float4 f2 = *(const float4*)(arow + k0 + 16);
        float4 f3 = *(const float4*)(arow + k0 + 20);
        __builtin_prefetch(arow + k0 + 32, 0, 0);   // global_prefetch_b8
        AFrag A;
        A.h[0]  = (_Float16)f0.x; A.h[1]  = (_Float16)f0.y;
        A.h[2]  = (_Float16)f0.z; A.h[3]  = (_Float16)f0.w;
        A.h[4]  = (_Float16)f1.x; A.h[5]  = (_Float16)f1.y;
        A.h[6]  = (_Float16)f1.z; A.h[7]  = (_Float16)f1.w;
        A.h[8]  = (_Float16)f2.x; A.h[9]  = (_Float16)f2.y;
        A.h[10] = (_Float16)f2.z; A.h[11] = (_Float16)f2.w;
        A.h[12] = (_Float16)f3.x; A.h[13] = (_Float16)f3.y;
        A.h[14] = (_Float16)f3.z; A.h[15] = (_Float16)f3.w;
#pragma unroll
        for (int nt = 0; nt < 4; nt++) {
            v16h Bv = *(const v16h*)(Bl + ((size_t)(nt * 24 + kc) * 32 + lane) * 16);
            acc[nt].v = __builtin_amdgcn_wmma_f32_16x16x32_f16(
                false, A.v, false, Bv, (short)0, acc[nt].v, false, false);
        }
    }
    __syncthreads();   // B buffer dead; safe to overlay zpre/zh

#pragma unroll
    for (int nt = 0; nt < 4; nt++)
#pragma unroll
        for (int j = 0; j < 8; j++)
            zp[(mb + lh * 8 + j) * 65 + nt * 16 + lm] = acc[nt].f[j];
    __syncthreads();

    // ---- LayerNorm: threads 0..127, one row each ----
    if (t < 128) {
        const float* sr = seq_in + (size_t)(rowbase + t) * DD;
        float mean = 0.0f;
        for (int c = 0; c < DD; c++) {
            float v = zp[t * 65 + c] + linb[c] + sr[c];
            zp[t * 65 + c] = v;
            mean += v;
        }
        mean *= (1.0f / DD);
        float var = 0.0f;
        for (int c = 0; c < DD; c++) { float d = zp[t * 65 + c] - mean; var += d * d; }
        var *= (1.0f / DD);
        float rstd = rsqrtf(var + 1e-3f);
        for (int c = 0; c < DD; c++)
            zhp[t * 72 + c] = (_Float16)(lng[c] * (zp[t * 65 + c] - mean) * rstd + lnb[c]);
    }
    __syncthreads();

    // ---- xp GEMMs: z[16x64] @ K[64x96] for both directions ----
    for (int dir = 0; dir < 2; dir++) {
        const _Float16* kp = dir ? kbp : kfp;
        const float*    bi = dir ? bib : bif;
        float*          xp = dir ? xpb : xpf;
#pragma unroll
        for (int nt = 0; nt < 6; nt++) {
            CFrag c2;
#pragma unroll
            for (int j = 0; j < 8; j++) c2.f[j] = 0.0f;
#pragma unroll
            for (int kc = 0; kc < 2; kc++) {
                const int k0 = kc * 32 + lh * 8;
                AFrag A;
                A.p[0] = *(const v8h*)&zhp[(mb + lm) * 72 + k0];
                A.p[1] = *(const v8h*)&zhp[(mb + lm) * 72 + k0 + 16];
                v16h Bv = *(const v16h*)(kp + ((size_t)(nt * 2 + kc) * 32 + lane) * 16);
                c2.v = __builtin_amdgcn_wmma_f32_16x16x32_f16(
                    false, A.v, false, Bv, (short)0, c2.v, false, false);
            }
            const int n = nt * 16 + lm;
            const float bv = bi[n];
#pragma unroll
            for (int j = 0; j < 8; j++) {
                int r = rowbase + mb + lh * 8 + j;
                xp[(size_t)r * 96 + n] = c2.f[j] + bv;
            }
        }
    }
}

// ---------------------------------------------------------------
// GRU recurrence: one block per direction; state [64 batches x 32] in LDS.
// R weights async-staged to LDS once (ASYNCcnt path). Each step: hp = H @ R
// (WMMA, [64,32]@[32,96], 24 tiles over 8 waves), then gate math.
__global__ __launch_bounds__(256)
void gru_kernel(const float* __restrict__ xp_f, const float* __restrict__ xp_b,
                const _Float16* __restrict__ rfp, const _Float16* __restrict__ rbp,
                const float* __restrict__ brf, const float* __restrict__ brb,
                float* __restrict__ seq_out, float* __restrict__ hF, float* __restrict__ hB) {
    const int dir = blockIdx.x;
    const float*    xp = dir ? xp_b : xp_f;
    const _Float16* rp = dir ? rbp : rfp;
    const float*    br = dir ? brb : brf;
    float*          hG = dir ? hB : hF;

    __shared__ float    Hs[64][33];
    __shared__ _Float16 Hh[64][40];              // 80 B / row, 16B-aligned
    __shared__ float    hp[64][97];
    __shared__ __align__(32) _Float16 Rl[3072];  // packed R weights (6KB)

    const int t    = threadIdx.x;
    const int w    = t >> 5;
    const int lane = t & 31;
    const int lm   = lane & 15;
    const int lh   = lane >> 4;

    // async-stage packed R into LDS: 6144 B = 384 x b128
    {
        const unsigned char* src = (const unsigned char*)rp;
        unsigned char* dst = (unsigned char*)Rl;
        for (int i = t; i < 384; i += 256)
            async_b128(src + (size_t)i * 16, dst + (size_t)i * 16);
        wait_async0();
    }
    for (int i = t; i < 64 * 32; i += 256) Hs[i >> 5][i & 31] = hG[i];
    __syncthreads();

    for (int step = 0; step < TT; step++) {
        const int tt = dir ? (TT - 1 - step) : step;

        for (int i = t; i < 64 * 32; i += 256)
            Hh[i >> 5][i & 31] = (_Float16)Hs[i >> 5][i & 31];
        __syncthreads();

#pragma unroll
        for (int q = 0; q < 3; q++) {
            const int job = w * 3 + q;
            const int mt = job / 6, nt = job % 6;
            AFrag A;
            A.p[0] = *(const v8h*)&Hh[mt * 16 + lm][lh * 8];
            A.p[1] = *(const v8h*)&Hh[mt * 16 + lm][16 + lh * 8];
            v16h Bv = *(const v16h*)(Rl + ((size_t)nt * 32 + lane) * 16);
            CFrag c2;
#pragma unroll
            for (int j = 0; j < 8; j++) c2.f[j] = 0.0f;
            c2.v = __builtin_amdgcn_wmma_f32_16x16x32_f16(
                false, A.v, false, Bv, (short)0, c2.v, false, false);
#pragma unroll
            for (int j = 0; j < 8; j++)
                hp[mt * 16 + lh * 8 + j][nt * 16 + lm] = c2.f[j];
        }
        __syncthreads();

        for (int i = t; i < 64 * 32; i += 256) {
            const int b = i >> 5, u = i & 31;
            const float* xr = xp + (size_t)(b * TT + tt) * 96;
            float xz = xr[u], xrr = xr[32 + u], xh = xr[64 + u];
            float hz = hp[b][u] + br[u];
            float hr = hp[b][32 + u] + br[32 + u];
            float hh = hp[b][64 + u] + br[64 + u];
            float z  = 1.0f / (1.0f + expf(-(xz + hz)));
            float r  = 1.0f / (1.0f + expf(-(xrr + hr)));
            float hc = tanhf(xh + r * hh);
            float h  = Hs[b][u];
            float hn = z * h + (1.0f - z) * hc;
            Hs[b][u] = hn;
            seq_out[(size_t)(b * TT + tt) * DD + dir * UU + u] = hn;
        }
        __syncthreads();
    }

    for (int i = t; i < 64 * 32; i += 256) hG[i] = Hs[i >> 5][i & 31];
}

// ---------------------------------------------------------------
// Build lower-triangular L (130x130) with softplus(diag)+1e-5 on the diagonal.
__global__ void build_L(const float* __restrict__ tril, float* __restrict__ L) {
    int idx = blockIdx.x * blockDim.x + threadIdx.x;
    if (idx >= NN * NN) return;
    int r = idx / NN, c = idx % NN;
    float v = 0.0f;
    if (c < r) {
        v = tril[r * (r + 1) / 2 + c];
    } else if (c == r) {
        float d = tril[r * (r + 1) / 2 + r];
        v = log1pf(expf(d)) + 1e-5f;
    }
    L[idx] = v;
}

// ---------------------------------------------------------------
// MC head: one block per sample s. w = post_loc + eps[s] @ L.T ;
// logits[b,c] = sum_d x[b,d] * w[2d+c] + w[128+c]; softmax; atomic accumulate.
__global__ __launch_bounds__(256)
void mc_kernel(const float* __restrict__ eps, const float* __restrict__ L,
               const float* __restrict__ post_loc,
               const float* __restrict__ hF, const float* __restrict__ hB,
               float* __restrict__ acc) {
    const int s = blockIdx.x;
    const int t = threadIdx.x;
    __shared__ float wv[NN];
    __shared__ float x[64][65];

    for (int n = t; n < NN; n += blockDim.x) {
        float sum = post_loc[n];
        const float* er = eps + (size_t)s * NN;
        const float* Lr = L + (size_t)n * NN;
        for (int j = 0; j <= n; j++) sum += er[j] * Lr[j];
        wv[n] = sum;
    }
    for (int i = t; i < 64 * 32; i += blockDim.x) {
        int b = i >> 5, u = i & 31;
        x[b][2 * u]     = hF[i];
        x[b][2 * u + 1] = hB[i];
    }
    __syncthreads();

    if (t < 64) {
        const int b = t;
        float l0 = wv[128], l1 = wv[129];
        for (int d = 0; d < DD; d++) {
            float xv = x[b][d];
            l0 += xv * wv[2 * d];
            l1 += xv * wv[2 * d + 1];
        }
        float m = fmaxf(l0, l1);
        float e0 = expf(l0 - m), e1 = expf(l1 - m);
        float inv = 1.0f / (e0 + e1);
        atomicAdd(&acc[2 * b],     e0 * inv);
        atomicAdd(&acc[2 * b + 1], e1 * inv);
    }
}

__global__ void finalize_kernel(const float* __restrict__ acc, float* __restrict__ out) {
    int i = threadIdx.x;
    if (i < BB * 2) out[i] = acc[i] * (1.0f / (float)SMC);
}

// ---------------------------------------------------------------
extern "C" void kernel_launch(void* const* d_in, const int* in_sizes, int n_in,
                              void* d_out, int out_size, void* d_ws, size_t ws_size,
                              hipStream_t stream) {
    (void)in_sizes; (void)n_in; (void)out_size; (void)ws_size;

    const float* hs       = (const float*)d_in[0];
    const float* lin_w    = (const float*)d_in[1];
    const float* lin_b    = (const float*)d_in[2];
    const float* gru_kf   = (const float*)d_in[3];
    const float* gru_rf   = (const float*)d_in[4];
    const float* gru_bf   = (const float*)d_in[5];
    const float* gru_kb   = (const float*)d_in[6];
    const float* gru_rb   = (const float*)d_in[7];
    const float* gru_bb   = (const float*)d_in[8];
    const float* ln_g     = (const float*)d_in[9];
    const float* ln_bv    = (const float*)d_in[10];
    const float* post_loc = (const float*)d_in[11];
    const float* post_tril= (const float*)d_in[12];
    const float* eps      = (const float*)d_in[13];
    float* out = (float*)d_out;

    // workspace layout (all region sizes multiples of 64 floats)
    float* seq  = (float*)d_ws;               // [8192*64]
    float* xpf  = seq + 524288;               // [8192*96]
    float* xpb  = xpf + 786432;               // [8192*96]
    float* hF   = xpb + 786432;               // [64*32]
    float* hB   = hF + 2048;                  // [64*32]
    float* Lm   = hB + 2048;                  // 130*130 (padded to 16960)
    float* accp = Lm + 16960;                 // [128]
    _Float16* lwp = (_Float16*)(accp + 128);  // 4*24*32*16 = 49152 halfs
    _Float16* kfp = lwp + 49152;              // 6*2*32*16  = 6144
    _Float16* kbp = kfp + 6144;
    _Float16* rfp = kbp + 6144;               // 6*1*32*16  = 3072
    _Float16* rbp = rfp + 3072;

    // init state
    zero_f32<<<(524288 + 255) / 256, 256, 0, stream>>>(seq, 524288);
    zero_f32<<<(4096   + 255) / 256, 256, 0, stream>>>(hF, 4096);      // hF + hB
    zero_f32<<<1, 256, 0, stream>>>(accp, 128);

    // pack weights into WMMA B-fragment layout
    pack_b_f16<<<(3072 + 255) / 256, 256, 0, stream>>>(lin_w,  lwp, 64, 24, 3072);
    pack_b_f16<<<2, 256, 0, stream>>>(gru_kf, kfp, 96, 2, 384);
    pack_b_f16<<<2, 256, 0, stream>>>(gru_kb, kbp, 96, 2, 384);
    pack_b_f16<<<1, 256, 0, stream>>>(gru_rf, rfp, 96, 1, 192);
    pack_b_f16<<<1, 256, 0, stream>>>(gru_rb, rbp, 96, 1, 192);

    // 13 stacked bidirectional GRU layers
    for (int i = 0; i < NUM_HS; i++) {
        layer_kernel<<<64, 256, 0, stream>>>(
            hs + (size_t)i * BB * TT * HH, seq, lwp, kfp, kbp,
            lin_b, gru_bf, gru_bb, ln_g, ln_bv, xpf, xpb);
        gru_kernel<<<2, 256, 0, stream>>>(
            xpf, xpb, rfp, rbp, gru_bf + 96, gru_bb + 96, seq, hF, hB);
    }

    // posterior MC head
    build_L<<<(NN * NN + 255) / 256, 256, 0, stream>>>(post_tril, Lm);
    mc_kernel<<<SMC, 256, 0, stream>>>(eps, Lm, post_loc, hF, hB, accp);
    finalize_kernel<<<1, 128, 0, stream>>>(accp, out);
}
